// RelationalTransformer_77017353551926
// MI455X (gfx1250) — matrix operand order repaired and checked by reading, not compiled, not run on hardware
//
#include <hip/hip_runtime.h>
#include <cstdint>

typedef __attribute__((ext_vector_type(16))) _Float16 v16h;
typedef __attribute__((ext_vector_type(8)))  _Float16 v8h;
typedef __attribute__((ext_vector_type(8)))  float    v8f;
typedef __attribute__((ext_vector_type(4)))  float    v4f;
typedef __attribute__((ext_vector_type(4)))  int      v4i;

#define NUM_DTYPES 5
#define HID 256
#define NH  8
#define HD  32
#define NL  12
#define MLPD 1024
#define VD  384
#define SD  384
#define TD  5
#define RD  64
#define BB  2
#define SS  512
#define MTOK (BB*SS)   // 1024 tokens

// Async global->LDS path (gfx1250): guarded so compile never breaks.
#if defined(__HIP_DEVICE_COMPILE__) && \
    __has_builtin(__builtin_amdgcn_global_load_async_to_lds_b128) && \
    __has_builtin(__builtin_amdgcn_s_wait_asynccnt)
#define GEMM_ASYNC 1
typedef __attribute__((address_space(1))) v4i* g_v4i_p;
typedef __attribute__((address_space(3))) v4i* l_v4i_p;
__device__ __forceinline__ void async_fill16(_Float16* dst, const _Float16* src) {
    __builtin_amdgcn_global_load_async_to_lds_b128((g_v4i_p)src, (l_v4i_p)dst, 0, 0);
}
#define WAIT_ASYNC() __builtin_amdgcn_s_wait_asynccnt(0)
#endif

#define SHUF16(a, b) __builtin_shufflevector(a, b, 0,1,2,3,4,5,6,7,8,9,10,11,12,13,14,15)

// ---------------------------------------------------------------------------
// Weight conversion: fp32 row-major [K,N] -> f16 swizzled into WMMA B-frag
// ---------------------------------------------------------------------------
__global__ void conv_w_kernel(const float* __restrict__ W, _Float16* __restrict__ out,
                              int NMAT, int K, int N) {
    long long idx = (long long)blockIdx.x * blockDim.x + threadIdx.x;
    long long KN = (long long)K * N;
    if (idx >= (long long)NMAT * KN) return;
    int  mat = (int)(idx / KN);
    long long rem = idx - (long long)mat * KN;
    int h    = (int)(rem & 15);
    int lane = (int)((rem >> 4) & 31);
    long long tile = rem >> 9;
    int ntiles = N >> 4;
    int nt = (int)(tile % ntiles);
    int kt = (int)(tile / ntiles);
    int srow = kt * 32 + (lane >> 4) * 8 + (h & 7) + (h >> 3) * 16;
    int scol = nt * 16 + (lane & 15);
    out[idx] = (_Float16)W[(long long)mat * KN + (long long)srow * N + scol];
}

// ---------------------------------------------------------------------------
// Mask prep (once per launch, not per layer): additive bias 0 / -1e30
// bias[b][q][k] = ((mask && pad[q] && pad[k]) || q==k) ? 0 : -1e30  (branchless)
// ---------------------------------------------------------------------------
__global__ void maskprep_kernel(const int* __restrict__ mask, const int* __restrict__ pad,
                                float* __restrict__ bias) {
    int idx = blockIdx.x * 256 + threadIdx.x;      // B*S*S = 524288
    int k = idx & 511, q = (idx >> 9) & 511, b = idx >> 18;
    int ok = (((mask[idx] != 0) & (pad[b * SS + q] != 0)) & (pad[b * SS + k] != 0)) | (q == k);
    bias[idx] = ok ? 0.f : -1e30f;
}

// ---------------------------------------------------------------------------
// Embedding / gating: one block (256 thr) per token
// ---------------------------------------------------------------------------
__global__ void embed_kernel(const float* __restrict__ val_e, const float* __restrict__ sch_e,
                             const float* __restrict__ tmp_e, const float* __restrict__ rol_e,
                             const float* __restrict__ W_schema, const float* __restrict__ W_temporal,
                             const float* __restrict__ W_role, const float* __restrict__ E_dtype,
                             const float* __restrict__ W_value, const float* __restrict__ W_gate,
                             const float* __restrict__ b_gate, const float* __restrict__ E_mask,
                             const int* __restrict__ dtype_ids, const int* __restrict__ is_masked,
                             const int* __restrict__ pad, float* __restrict__ x) {
    __shared__ float ve[VD], se[SD], te[TD], re[RD], ctx[5 * HID];
    int tok = blockIdx.x;
    int j = threadIdx.x;
    for (int k = j; k < VD; k += 256) { ve[k] = val_e[(long long)tok * VD + k]; se[k] = sch_e[(long long)tok * SD + k]; }
    if (j < TD) te[j] = tmp_e[tok * TD + j];
    if (j < RD) re[j] = rol_e[tok * RD + j];
    __syncthreads();
    int d = dtype_ids[tok];
    int mk = is_masked[tok];
    int pv = pad[tok];
    float vp;
    if (mk) {
        vp = E_mask[d * HID + j];
    } else {
        vp = 0.f;
        const float* wv = W_value + (long long)d * VD * HID + j;
        for (int k = 0; k < VD; ++k) vp += ve[k] * wv[(long long)k * HID];
    }
    float sp = 0.f;
    for (int k = 0; k < SD; ++k) sp += se[k] * W_schema[(long long)k * HID + j];
    float tp = 0.f;
    for (int k = 0; k < TD; ++k) tp += te[k] * W_temporal[k * HID + j];
    float rp = 0.f;
    for (int k = 0; k < RD; ++k) rp += re[k] * W_role[k * HID + j];
    float de = E_dtype[d * HID + j];
    ctx[j] = vp; ctx[HID + j] = sp; ctx[2 * HID + j] = tp; ctx[3 * HID + j] = rp; ctx[4 * HID + j] = de;
    __syncthreads();
    float g = b_gate[j];
    for (int k = 0; k < 5 * HID; ++k) g += ctx[k] * W_gate[(long long)k * HID + j];
    float gate = 1.f / (1.f + __expf(-g));
    x[(long long)tok * HID + j] = (vp + sp + tp + rp + de) * gate * (pv ? 1.f : 0.f);
}

// ---------------------------------------------------------------------------
// LayerNorm: one wave per token (lane handles 8 channels)
// ---------------------------------------------------------------------------
__global__ void ln_kernel(const float* __restrict__ x, const float* __restrict__ g,
                          const float* __restrict__ b, _Float16* __restrict__ o16,
                          float* __restrict__ o32) {
    int tok  = blockIdx.x * 8 + (threadIdx.x >> 5);
    int lane = threadIdx.x & 31;
    const float* xp = x + (long long)tok * HID + lane * 8;
    float v[8]; float s1 = 0.f, s2 = 0.f;
#pragma unroll
    for (int i = 0; i < 8; ++i) { v[i] = xp[i]; s1 += v[i]; s2 += v[i] * v[i]; }
#pragma unroll
    for (int off = 16; off >= 1; off >>= 1) { s1 += __shfl_xor(s1, off, 32); s2 += __shfl_xor(s2, off, 32); }
    float m  = s1 * (1.f / HID);
    float vr = s2 * (1.f / HID) - m * m;
    float rs = rsqrtf(vr + 1e-5f);
#pragma unroll
    for (int i = 0; i < 8; ++i) {
        int c = lane * 8 + i;
        float y = (v[i] - m) * rs * g[c] + b[c];
        if (o16) o16[(long long)tok * HID + c] = (_Float16)y;
        if (o32) o32[(long long)tok * HID + c] = y;
    }
}

// ---------------------------------------------------------------------------
// WMMA GEMM, LDS-staged W panel with async double-buffering.
// block = 256 thr (8 waves); wave computes 16 x (NT*16) strip (NT accumulators).
// grid = (N/(16*NT), M/128)
// ---------------------------------------------------------------------------
template <int NT>
__global__ void gemm_wmma_kernel(const _Float16* __restrict__ A, const _Float16* __restrict__ Wsw,
                                 const float* __restrict__ bias, const float* __restrict__ resid,
                                 float* __restrict__ outF, _Float16* __restrict__ outH,
                                 int N, int K) {
    __shared__ _Float16 wlds[2][NT * 512];
    int tid  = threadIdx.x;
    int wave = tid >> 5;
    int lane = tid & 31;
    int nt0 = blockIdx.x * NT;
    int mt  = blockIdx.y * 8 + wave;
    int m0 = mt * 16, n0 = nt0 * 16;
    int row = m0 + (lane & 15);
    int kb  = (lane >> 4) * 8;
    int ntilesTot = N >> 4;
    int nk = K >> 5;
    bool filler = tid < NT * 64;                         // each filler thread moves 16B/panel
    const _Float16* panel0 = Wsw + (long long)nt0 * 512; // + kt*ntilesTot*512 (contiguous NT tiles)

    v8f acc[NT];
#pragma unroll
    for (int i = 0; i < NT; ++i) acc[i] = (v8f){};

#ifdef GEMM_ASYNC
    if (filler) async_fill16(&wlds[0][tid * 8], panel0 + tid * 8);
#endif
    for (int kt = 0; kt < nk; ++kt) {
        const _Float16* ap = A + (long long)row * K + (kt << 5) + kb;
        v8h a0 = *(const v8h*)ap;
        v8h a1 = *(const v8h*)(ap + 16);
        __builtin_prefetch(ap + 32, 0, 1);
        v16h av = SHUF16(a0, a1);
#ifdef GEMM_ASYNC
        WAIT_ASYNC();             // my share of buffer kt is in LDS
        __syncthreads();          // everyone's share is in LDS
        if (kt + 1 < nk && filler)
            async_fill16(&wlds[(kt + 1) & 1][tid * 8],
                         panel0 + (long long)(kt + 1) * ntilesTot * 512 + tid * 8);
        const _Float16* lb = &wlds[kt & 1][0];
#else
        v8h wreg;
        if (filler) wreg = *(const v8h*)(panel0 + (long long)kt * ntilesTot * 512 + tid * 8);
        __syncthreads();          // previous iteration's LDS reads complete
        if (filler) *(v8h*)&wlds[0][tid * 8] = wreg;
        __syncthreads();
        const _Float16* lb = &wlds[0][0];
#endif
#pragma unroll
        for (int i = 0; i < NT; ++i) {
            const _Float16* lp = lb + i * 512 + lane * 16;
            v8h b0 = *(const v8h*)lp;
            v8h b1 = *(const v8h*)(lp + 8);
            v16h bv = SHUF16(b0, b1);
            acc[i] = __builtin_amdgcn_wmma_f32_16x16x32_f16(false, av, false, bv, (short)0, acc[i], false, false);
        }
    }
    int rbase = m0 + ((lane >> 4) << 3);
#pragma unroll
    for (int i = 0; i < NT; ++i) {
        int col = n0 + i * 16 + (lane & 15);
        float bvv = bias ? bias[col] : 0.f;
#pragma unroll
        for (int r = 0; r < 8; ++r) {
            long long o = (long long)(rbase + r) * N + col;
            float v = acc[i][r] + bvv;
            if (resid) v += resid[o];
            if (outF) outF[o] = v;
            if (outH) outH[o] = (_Float16)v;
        }
    }
}

// ---------------------------------------------------------------------------
// Transpose V within qkv to d-major: vt[b][h][d][s]
// ---------------------------------------------------------------------------
__global__ void vtrans_kernel(const _Float16* __restrict__ qkv, _Float16* __restrict__ vt) {
    int idx = blockIdx.x * 256 + threadIdx.x;       // 2*8*32*512 = 262144
    int s = idx & 511, d = (idx >> 9) & 31, h = (idx >> 14) & 7, b = idx >> 17;
    vt[idx] = qkv[(long long)(b * SS + s) * 768 + 512 + h * HD + d];
}

// ---------------------------------------------------------------------------
// Fused masked attention for one (qtile, head, batch): 1 wave per block
// scores WMMA (+precomputed additive mask bias) -> softmax in LDS -> P*V WMMA
// ---------------------------------------------------------------------------
__global__ void attn_kernel(const _Float16* __restrict__ qkv, const _Float16* __restrict__ vt,
                            const float* __restrict__ bias, _Float16* __restrict__ ctx) {
    __shared__ float sc[16][512];
    int lane = threadIdx.x & 31;
    int qt = blockIdx.x, h = blockIdx.y, b = blockIdx.z;
    int q0 = qt * 16;
    int lrow = lane & 15;
    int half = lane >> 4;
    int kb = half * 16;
    // A fragment = Q tile (16 x 32), loaded once
    const _Float16* qp = qkv + (long long)(b * SS + q0 + lrow) * 768 + h * HD + kb;
    v8h qa0 = *(const v8h*)qp;
    v8h qa1 = *(const v8h*)(qp + 8);
    v16h qa = SHUF16(qa0, qa1);
    const float scale = 0.17677669529663687f;  // 1/sqrt(32)
    const float* brow = bias + ((long long)b * SS + q0) * SS;   // + q*SS + k
    for (int kt = 0; kt < 32; ++kt) {
        int kt0 = kt * 16;
        // B fragment = K^T tile: lane = key token, halves = head-dim (contiguous)
        const _Float16* kp = qkv + (long long)(b * SS + kt0 + lrow) * 768 + 256 + h * HD + half * 16;
        v8h kb0 = *(const v8h*)kp;
        v8h kb1 = *(const v8h*)(kp + 8);
        v16h kbv = SHUF16(kb0, kb1);
        v8f s = {};
        s = __builtin_amdgcn_wmma_f32_16x16x32_f16(false, qa, false, kbv, (short)0, s, false, false);
        int kk = kt0 + lrow;
#pragma unroll
        for (int r = 0; r < 8; ++r) {
            int qr = r + 8 * half;                      // row within tile
            sc[qr][kk] = s[r] * scale + brow[(long long)qr * SS + kk];
        }
    }
    __syncthreads();
    // softmax: lane pair (l, l^16) shares row l&15, each scans half the columns (float4)
    int c0 = half * 256;
    const v4f* scr = (const v4f*)&sc[lrow][c0];
    float mx = -3.0e38f;
    for (int c = 0; c < 64; ++c) {
        v4f t = scr[c];
        mx = fmaxf(mx, fmaxf(fmaxf(t.x, t.y), fmaxf(t.z, t.w)));
    }
    mx = fmaxf(mx, __shfl_xor(mx, 16, 32));
    float sum = 0.f;
    v4f* scw = (v4f*)&sc[lrow][c0];
    for (int c = 0; c < 64; ++c) {
        v4f t = scw[c];
        t.x = __expf(t.x - mx); t.y = __expf(t.y - mx);
        t.z = __expf(t.z - mx); t.w = __expf(t.w - mx);
        sum += t.x + t.y + t.z + t.w;
        scw[c] = t;
    }
    sum += __shfl_xor(sum, 16, 32);
    float inv = 1.f / sum;   // diag always unmasked -> sum > 0
    __syncthreads();
    // O = P @ V (fold 1/sum into A-frag conversion); float4 LDS reads
    v8f o0 = {}, o1 = {};
    for (int kt = 0; kt < 16; ++kt) {
        int k0 = kt * 32;
        const v4f* p0 = (const v4f*)&sc[lrow][k0 + kb];
        const v4f* p1 = (const v4f*)&sc[lrow][k0 + kb + 8];
        v4f t0 = p0[0], t1 = p0[1], t2 = p1[0], t3 = p1[1];
        v16h wa;
        wa[0]  = (_Float16)(t0.x * inv); wa[1]  = (_Float16)(t0.y * inv);
        wa[2]  = (_Float16)(t0.z * inv); wa[3]  = (_Float16)(t0.w * inv);
        wa[4]  = (_Float16)(t1.x * inv); wa[5]  = (_Float16)(t1.y * inv);
        wa[6]  = (_Float16)(t1.z * inv); wa[7]  = (_Float16)(t1.w * inv);
        wa[8]  = (_Float16)(t2.x * inv); wa[9]  = (_Float16)(t2.y * inv);
        wa[10] = (_Float16)(t2.z * inv); wa[11] = (_Float16)(t2.w * inv);
        wa[12] = (_Float16)(t3.x * inv); wa[13] = (_Float16)(t3.y * inv);
        wa[14] = (_Float16)(t3.z * inv); wa[15] = (_Float16)(t3.w * inv);
        const _Float16* vp0 = vt + ((long long)((b * NH + h) * HD + lrow) * SS) + k0 + half * 16;
        const _Float16* vp1 = vt + ((long long)((b * NH + h) * HD + 16 + lrow) * SS) + k0 + half * 16;
        v8h b00 = *(const v8h*)vp0; v8h b01 = *(const v8h*)(vp0 + 8);
        v8h b10 = *(const v8h*)vp1; v8h b11 = *(const v8h*)(vp1 + 8);
        v16h bv0 = SHUF16(b00, b01);
        v16h bv1 = SHUF16(b10, b11);
        o0 = __builtin_amdgcn_wmma_f32_16x16x32_f16(false, wa, false, bv0, (short)0, o0, false, false);
        o1 = __builtin_amdgcn_wmma_f32_16x16x32_f16(false, wa, false, bv1, (short)0, o1, false, false);
    }
#pragma unroll
    for (int r = 0; r < 8; ++r) {
        long long row = (long long)(b * SS + q0 + r + 8 * half) * HID + h * HD;
        ctx[row + lrow]      = (_Float16)o0[r];
        ctx[row + 16 + lrow] = (_Float16)o1[r];
    }
}

// ---------------------------------------------------------------------------
// SwiGLU: h2 = silu(h1[:, :1024]) * h1[:, 1024:]  -> f16
// ---------------------------------------------------------------------------
__global__ void swiglu_kernel(const float* __restrict__ h1, _Float16* __restrict__ h2) {
    int idx = blockIdx.x * 256 + threadIdx.x;   // 1024*1024
    int row = idx >> 10, c = idx & 1023;
    float g = h1[(long long)row * 2048 + c];
    float a = h1[(long long)row * 2048 + 1024 + c];
    h2[idx] = (_Float16)((g / (1.f + __expf(-g))) * a);
}

// ---------------------------------------------------------------------------
extern "C" void kernel_launch(void* const* d_in, const int* in_sizes, int n_in,
                              void* d_out, int out_size, void* d_ws, size_t ws_size,
                              hipStream_t stream) {
    (void)in_sizes; (void)n_in; (void)out_size; (void)ws_size;
    const float* val_e   = (const float*)d_in[0];
    const float* sch_e   = (const float*)d_in[1];
    const float* tmp_e   = (const float*)d_in[2];
    const float* rol_e   = (const float*)d_in[3];
    const float* W_schema= (const float*)d_in[4];
    const float* W_temporal=(const float*)d_in[5];
    const float* W_role  = (const float*)d_in[6];
    const float* E_dtype = (const float*)d_in[7];
    const float* W_value = (const float*)d_in[8];
    const float* W_gate  = (const float*)d_in[9];
    const float* b_gate  = (const float*)d_in[10];
    const float* E_mask  = (const float*)d_in[11];
    const float* qkv_w   = (const float*)d_in[12];
    const float* out_w   = (const float*)d_in[13];
    const float* ln_g    = (const float*)d_in[14];
    const float* ln_b    = (const float*)d_in[15];
    const float* fc1_w   = (const float*)d_in[16];
    const float* fc1_b   = (const float*)d_in[17];
    const float* fc2_w   = (const float*)d_in[18];
    const float* fc2_b   = (const float*)d_in[19];
    const float* lnf_g   = (const float*)d_in[20];
    const float* lnf_b   = (const float*)d_in[21];
    const int*   dtype_ids = (const int*)d_in[22];
    const int*   masks[5]  = { (const int*)d_in[23], (const int*)d_in[24], (const int*)d_in[25],
                               (const int*)d_in[26], (const int*)d_in[27] };
    const int*   is_masked = (const int*)d_in[28];
    const int*   pad       = (const int*)d_in[29];

    // workspace carve-up (256B aligned)
    char* ws = (char*)d_ws;
    size_t off = 0;
    auto carve = [&](size_t bytes) { size_t o = off; off = (off + bytes + 255) & ~(size_t)255; return o; };
    _Float16* wsQKV = (_Float16*)(ws + carve((size_t)NL * 5 * 256 * 768 * 2));
    _Float16* wsOUT = (_Float16*)(ws + carve((size_t)NL * 5 * 256 * 256 * 2));
    _Float16* wsF1  = (_Float16*)(ws + carve((size_t)NL * 256 * 2048 * 2));
    _Float16* wsF2  = (_Float16*)(ws + carve((size_t)NL * 1024 * 256 * 2));
    float*    bias5 = (float*)   (ws + carve((size_t)5 * BB * SS * SS * 4));
    float*    x     = (float*)   (ws + carve((size_t)MTOK * HID * 4));
    _Float16* ln16  = (_Float16*)(ws + carve((size_t)MTOK * HID * 2));
    _Float16* qkv16 = (_Float16*)(ws + carve((size_t)MTOK * 768 * 2));
    _Float16* vt16  = (_Float16*)(ws + carve((size_t)BB * NH * HD * SS * 2));
    _Float16* ctx16 = (_Float16*)(ws + carve((size_t)MTOK * HID * 2));
    float*    h1    = (float*)   (ws + carve((size_t)MTOK * 2048 * 4));
    _Float16* h2    = (_Float16*)(ws + carve((size_t)MTOK * 1024 * 2));

    // convert + swizzle all weights to f16 WMMA B-fragment layout
    auto conv = [&](const float* w, _Float16* o, int nmat, int K, int N) {
        long long tot = (long long)nmat * K * N;
        conv_w_kernel<<<dim3((unsigned)((tot + 255) / 256)), 256, 0, stream>>>(w, o, nmat, K, N);
    };
    conv(qkv_w, wsQKV, NL * 5, 256, 768);
    conv(out_w, wsOUT, NL * 5, 256, 256);
    conv(fc1_w, wsF1,  NL,     256, 2048);
    conv(fc2_w, wsF2,  NL,    1024, 256);

    // prep all 5 attention masks once (reused across all 12 layers)
    for (int a = 0; a < 5; ++a)
        maskprep_kernel<<<(BB * SS * SS) / 256, 256, 0, stream>>>(
            masks[a], pad, bias5 + (size_t)a * BB * SS * SS);

    embed_kernel<<<MTOK, 256, 0, stream>>>(val_e, sch_e, tmp_e, rol_e, W_schema, W_temporal,
                                           W_role, E_dtype, W_value, W_gate, b_gate, E_mask,
                                           dtype_ids, is_masked, pad, x);

    for (int l = 0; l < NL; ++l) {
        for (int a = 0; a < 5; ++a) {
            ln_kernel<<<MTOK / 8, 256, 0, stream>>>(x, ln_g + (l * 6 + a) * HID, ln_b + (l * 6 + a) * HID,
                                                    ln16, nullptr);
            gemm_wmma_kernel<4><<<dim3(768 / 64, MTOK / 128), 256, 0, stream>>>(
                ln16, wsQKV + (size_t)(l * 5 + a) * 256 * 768, nullptr, nullptr, nullptr, qkv16, 768, 256);
            vtrans_kernel<<<(BB * NH * HD * SS) / 256, 256, 0, stream>>>(qkv16, vt16);
            attn_kernel<<<dim3(SS / 16, NH, BB), 32, 0, stream>>>(
                qkv16, vt16, bias5 + (size_t)a * BB * SS * SS, ctx16);
            gemm_wmma_kernel<2><<<dim3(HID / 32, MTOK / 128), 256, 0, stream>>>(
                ctx16, wsOUT + (size_t)(l * 5 + a) * 256 * 256, nullptr, x, x, nullptr, HID, 256);
        }
        ln_kernel<<<MTOK / 8, 256, 0, stream>>>(x, ln_g + (l * 6 + 5) * HID, ln_b + (l * 6 + 5) * HID,
                                                ln16, nullptr);
        gemm_wmma_kernel<4><<<dim3(2048 / 64, MTOK / 128), 256, 0, stream>>>(
            ln16, wsF1 + (size_t)l * 256 * 2048, fc1_b + l * 2048, nullptr, h1, nullptr, 2048, 256);
        swiglu_kernel<<<(MTOK * 1024) / 256, 256, 0, stream>>>(h1, h2);
        gemm_wmma_kernel<2><<<dim3(HID / 32, MTOK / 128), 256, 0, stream>>>(
            h2, wsF2 + (size_t)l * 1024 * 256, fc2_b + l * HID, x, x, nullptr, HID, 1024);
    }
    ln_kernel<<<MTOK / 8, 256, 0, stream>>>(x, lnf_g, lnf_b, nullptr, (float*)d_out);
}
